// GroupBimodalCSRPool_2233382994384
// MI455X (gfx1250) — matrix operand order repaired and checked by reading, not compiled, not run on hardware
//
#include <hip/hip_runtime.h>
#include <math.h>

// ---------------------------------------------------------------------------
// types / helpers
// ---------------------------------------------------------------------------
typedef __bf16 v16bf __attribute__((ext_vector_type(16)));
typedef __bf16 v8bf  __attribute__((ext_vector_type(8)));
typedef float  v8f   __attribute__((ext_vector_type(8)));
typedef int    i32x4 __attribute__((ext_vector_type(4)));
typedef int    i32x8 __attribute__((ext_vector_type(8)));

__device__ __forceinline__ unsigned short f32_to_bf16(float f) {
  unsigned int u = __builtin_bit_cast(unsigned int, f);
  u += 0x7fffu + ((u >> 16) & 1u);          // round-to-nearest-even
  return (unsigned short)(u >> 16);
}
__device__ __forceinline__ float bf16_to_f32(unsigned short s) {
  unsigned int u = ((unsigned int)s) << 16;
  return __builtin_bit_cast(float, u);
}

// ---------------------------------------------------------------------------
// Kernel 0: convert W_mod1 / W_mod2 (128x128 f32, k-major) to bf16 workspace,
// TRANSPOSED to [n][k] so B-fragments are contiguous in LDS.
// ---------------------------------------------------------------------------
__global__ __launch_bounds__(256) void k_wcvt(const float* __restrict__ W1,
                                              const float* __restrict__ W2,
                                              unsigned short* __restrict__ o1t,
                                              unsigned short* __restrict__ o2t) {
  int i = blockIdx.x * 256 + threadIdx.x;   // 16384 elements each
  if (i < 128 * 128) {
    int k = i >> 7, n = i & 127;
    o1t[n * 128 + k] = f32_to_bf16(W1[i]);
    o2t[n * 128 + k] = f32_to_bf16(W2[i]);
  }
}

// ---------------------------------------------------------------------------
// Kernel 1: h = relu(relu(x_map @ W_e1a) @ W_e1b)   [V,8] -> [V,32]
// ---------------------------------------------------------------------------
__global__ __launch_bounds__(256) void k_enc1(const float* __restrict__ x_map,
                                              const float* __restrict__ W_e1a,
                                              const float* __restrict__ W_e1b,
                                              float* __restrict__ h, int V) {
  __shared__ float sA[8 * 32];
  __shared__ float sB[32 * 32];
  int t = threadIdx.x;
  if (t < 8 * 32) sA[t] = W_e1a[t];
  for (int i = t; i < 32 * 32; i += 256) sB[i] = W_e1b[i];
  __syncthreads();

  int v = blockIdx.x * 256 + t;
  if (v >= V) return;

  float xm[8];
#pragma unroll
  for (int i = 0; i < 8; ++i) xm[i] = x_map[(size_t)v * 8 + i];

  float t1[32];
#pragma unroll
  for (int c = 0; c < 32; ++c) {
    float s = 0.f;
#pragma unroll
    for (int i = 0; i < 8; ++i) s += xm[i] * sA[i * 32 + c];
    t1[c] = fmaxf(s, 0.f);
  }
#pragma unroll
  for (int c = 0; c < 32; ++c) {
    float s = 0.f;
#pragma unroll
    for (int i = 0; i < 32; ++i) s += t1[i] * sB[i * 32 + c];
    h[(size_t)v * 32 + c] = fmaxf(s, 0.f);
  }
}

// ---------------------------------------------------------------------------
// Kernel 2 (WMMA + TDM): x_mod_e = relu(relu(x_mod @ W1) @ W2), bf16 out.
// 128-row tile per workgroup, 8 waves, wave owns a 16-row strip.
// sIn: row-major [m][k] bf16.  sWt: TRANSPOSED [n][k] bf16.
// All fragment loads are 16-byte ds_load_b128; two output tiles in flight to
// cover the WMMA->VALU data hazard (ISA 7.12.1: 4 co-exec slots for bf16).
// ---------------------------------------------------------------------------
__device__ __forceinline__ v8f wmma_tile_128(const __bf16* __restrict__ sIn,
                                             const __bf16* __restrict__ sWt,
                                             int mrow, int hs, int ml, int nb) {
  v8f acc = {0.f, 0.f, 0.f, 0.f, 0.f, 0.f, 0.f, 0.f};
  const int ncol = (nb << 4) + ml;
#pragma unroll
  for (int kc = 0; kc < 4; ++kc) {
    const int K0 = kc << 5;
    // A: VGPRs 0-3 = K0+hs*8 .. +7 ; VGPRs 4-7 = K0+16+hs*8 .. +7
    v8bf alo = *(const v8bf*)(sIn + mrow * 128 + K0 + (hs << 3));
    v8bf ahi = *(const v8bf*)(sIn + mrow * 128 + K0 + 16 + (hs << 3));
    v16bf a = __builtin_shufflevector(alo, ahi, 0, 1, 2, 3, 4, 5, 6, 7,
                                      8, 9, 10, 11, 12, 13, 14, 15);
    // B (lane = column ncol): VGPRs 0-3 = K0+hs*8 .. +7 ; VGPRs 4-7 = K0+16+hs*8 .. +7
    v8bf blo = *(const v8bf*)(sWt + ncol * 128 + K0 + (hs << 3));
    v8bf bhi = *(const v8bf*)(sWt + ncol * 128 + K0 + 16 + (hs << 3));
    v16bf b = __builtin_shufflevector(blo, bhi, 0, 1, 2, 3, 4, 5, 6, 7,
                                      8, 9, 10, 11, 12, 13, 14, 15);
    acc = __builtin_amdgcn_wmma_f32_16x16x32_bf16(false, a, false, b,
                                                  (short)0, acc, false, false);
  }
  return acc;
}

// Issue one TDM copy: 32 KB contiguous global -> LDS (4096 x 8-byte elements).
// D# per ISA ch.8: group0 = {count=1, lds_addr, global_addr, type=2},
// group1 = {data_size=8B, tensor_dim0=tile_dim0=4096, stride0=4096}.
__device__ __forceinline__ void tdm_copy_32k(const void* gsrc, unsigned lds_off) {
  unsigned long long ga = (unsigned long long)(uintptr_t)gsrc;
  i32x4 g0;
  g0.x = 1;                                   // count=1, gather off
  g0.y = (int)lds_off;                        // lds_addr
  g0.z = (int)(ga & 0xffffffffu);             // global_addr[31:0]
  g0.w = (int)(((ga >> 32) & 0x1ffffffu) | (2u << 30)); // addr[56:32] | type=2
  i32x8 g1;
  g1[0] = 0x00030000;                         // wg_mask=0, data_size=8B
  g1[1] = 0x10000000;                         // tensor_dim0 = 4096 (low16<<16)
  g1[2] = 0;                                  // tensor_dim0 hi / tensor_dim1 lo
  g1[3] = 0x10000000;                         // tile_dim0 = 4096 (bits 127:112)
  g1[4] = 0;                                  // tile_dim1=0, tile_dim2=0
  g1[5] = 4096;                               // tensor_dim0_stride
  g1[6] = 0;
  g1[7] = 0;
  asm volatile("tensor_load_to_lds %0, %1" :: "s"(g0), "s"(g1) : "memory");
}

__global__ __launch_bounds__(256) void k_modmlp(const float* __restrict__ x_mod,
                                                const unsigned short* __restrict__ W1t,
                                                const unsigned short* __restrict__ W2t,
                                                unsigned short* __restrict__ xme,
                                                int V) {
  __shared__ __bf16 sW1[128 * 128];  // 32 KB, [n][k]
  __shared__ __bf16 sW2[128 * 128];  // 32 KB, [n][k]
  __shared__ __bf16 sX[128 * 128];   // 32 KB, [m][k]
  __shared__ __bf16 sY[128 * 128];   // 32 KB, [m][k]

  const int t = threadIdx.x;

  // --- stage both weight matrices via Tensor Data Mover (wave 0 only) ---
  if (t < 32) {
    tdm_copy_32k(W1t, (unsigned)(uintptr_t)&sW1[0]);
    tdm_copy_32k(W2t, (unsigned)(uintptr_t)&sW2[0]);
    __builtin_amdgcn_s_wait_tensorcnt(0);
  }

  // --- stage X tile, f32 -> bf16 (4096 float4 reads, uint2 LDS stores) ---
  const int base = blockIdx.x * 128;
#pragma unroll
  for (int i = 0; i < 16; ++i) {
    int idx4 = t + 256 * i;
    int row = idx4 >> 5;               // 32 float4 per row
    int col = (idx4 & 31) << 2;
    int v = base + row;
    float4 val = make_float4(0.f, 0.f, 0.f, 0.f);
    if (v < V) val = *(const float4*)(x_mod + (size_t)v * 128 + col);
    uint2 pk;
    pk.x = (unsigned)f32_to_bf16(val.x) | ((unsigned)f32_to_bf16(val.y) << 16);
    pk.y = (unsigned)f32_to_bf16(val.z) | ((unsigned)f32_to_bf16(val.w) << 16);
    *(uint2*)&sX[row * 128 + col] = pk;
  }
  __syncthreads();   // weights (TDM, waited by wave 0) + X tile visible to all

  const int lane = t & 31;
  const int wv   = t >> 5;
  const int mb   = wv << 4;            // wave's 16-row strip
  const int hs   = lane >> 4;
  const int ml   = lane & 15;
  const int mrow = mb + ml;

  // layer 1: sX @ W1 -> relu -> sY (bf16); two tiles in flight
  for (int nb = 0; nb < 8; nb += 2) {
    v8f acc0 = wmma_tile_128(sX, sW1, mrow, hs, ml, nb);
    v8f acc1 = wmma_tile_128(sX, sW1, mrow, hs, ml, nb + 1);
#pragma unroll
    for (int r = 0; r < 8; ++r) {
      int row = mb + (hs << 3) + r;
      sY[row * 128 + (nb << 4) + ml] =
          __builtin_bit_cast(__bf16, f32_to_bf16(fmaxf(acc0[r], 0.f)));
      sY[row * 128 + ((nb + 1) << 4) + ml] =
          __builtin_bit_cast(__bf16, f32_to_bf16(fmaxf(acc1[r], 0.f)));
    }
  }
  __syncthreads();

  // layer 2: sY @ W2 -> relu -> global (bf16); two tiles in flight
  for (int nb = 0; nb < 8; nb += 2) {
    v8f acc0 = wmma_tile_128(sY, sW2, mrow, hs, ml, nb);
    v8f acc1 = wmma_tile_128(sY, sW2, mrow, hs, ml, nb + 1);
#pragma unroll
    for (int r = 0; r < 8; ++r) {
      int vrow = base + mb + (hs << 3) + r;
      if (vrow < V) {
        xme[(size_t)vrow * 128 + (nb << 4) + ml] =
            f32_to_bf16(fmaxf(acc0[r], 0.f));
        xme[(size_t)vrow * 128 + ((nb + 1) << 4) + ml] =
            f32_to_bf16(fmaxf(acc1[r], 0.f));
      }
    }
  }
}

// ---------------------------------------------------------------------------
// Kernel 3: per-segment max over h, then set-MLP (32->32->32)
// ---------------------------------------------------------------------------
__global__ __launch_bounds__(256) void k_set(const int* __restrict__ csr,
                                             const float* __restrict__ h,
                                             const float* __restrict__ W_seta,
                                             const float* __restrict__ W_setb,
                                             float* __restrict__ hset, int N) {
  __shared__ float sA[32 * 32];
  __shared__ float sB[32 * 32];
  int t = threadIdx.x;
  for (int i = t; i < 32 * 32; i += 256) { sA[i] = W_seta[i]; sB[i] = W_setb[i]; }
  __syncthreads();

  int n = blockIdx.x * 256 + t;
  if (n >= N) return;
  int s = csr[n], e = csr[n + 1];

  float m[32];
  if (e <= s) {
#pragma unroll
    for (int c = 0; c < 32; ++c) hset[(size_t)n * 32 + c] = 0.f;
    return;
  }
#pragma unroll
  for (int c = 0; c < 32; ++c) m[c] = -3.4e38f;
  for (int p = s; p < e; ++p) {
    const float* hp = h + (size_t)p * 32;
#pragma unroll
    for (int c = 0; c < 32; ++c) m[c] = fmaxf(m[c], hp[c]);
  }
  float t1[32];
#pragma unroll
  for (int c = 0; c < 32; ++c) {
    float acc = 0.f;
#pragma unroll
    for (int i = 0; i < 32; ++i) acc += m[i] * sA[i * 32 + c];
    t1[c] = fmaxf(acc, 0.f);
  }
#pragma unroll
  for (int c = 0; c < 32; ++c) {
    float acc = 0.f;
#pragma unroll
    for (int i = 0; i < 32; ++i) acc += t1[i] * sB[i * 32 + c];
    hset[(size_t)n * 32 + c] = fmaxf(acc, 0.f);
  }
}

// ---------------------------------------------------------------------------
// Kernel 4: per-segment x_enc MLP -> scores -> scaled softmax -> pooled output
// ---------------------------------------------------------------------------
__global__ __launch_bounds__(256) void k_pool(const int* __restrict__ csr,
                                              const float* __restrict__ h,
                                              const float* __restrict__ hset,
                                              const unsigned short* __restrict__ xme,
                                              const float* __restrict__ W_e2a,
                                              const float* __restrict__ W_e2b,
                                              const float* __restrict__ W_score,
                                              const float* __restrict__ b_score,
                                              const float* __restrict__ g_w,
                                              const float* __restrict__ g_b,
                                              float* __restrict__ Cws,
                                              float* __restrict__ out, int N) {
  __shared__ float sA[64 * 32];   // W_e2a
  __shared__ float sB[32 * 32];   // W_e2b
  __shared__ float sS[32 * 16];   // W_score
  __shared__ float sbs[16], sgw[16], sgb[16];
  int t = threadIdx.x;
  for (int i = t; i < 64 * 32; i += 256) sA[i] = W_e2a[i];
  for (int i = t; i < 32 * 32; i += 256) sB[i] = W_e2b[i];
  for (int i = t; i < 32 * 16; i += 256) sS[i] = W_score[i];
  if (t < 16) { sbs[t] = b_score[t]; sgw[t] = g_w[t]; sgb[t] = g_b[t]; }
  __syncthreads();

  int n = blockIdx.x * 256 + t;
  if (n >= N) return;
  int s = csr[n], e = csr[n + 1];
  int cnt = e - s;
  float* orow = out + (size_t)n * 128;

  if (cnt <= 0) {
#pragma unroll 8
    for (int c = 0; c < 128; ++c) orow[c] = 0.f;
    out[(size_t)N * 128 + n] = 0.f;   // x_seen
    return;
  }
  out[(size_t)N * 128 + n] = 1.f;     // x_seen

  float hs[32];
#pragma unroll
  for (int c = 0; c < 32; ++c) hs[c] = hset[(size_t)n * 32 + c];

  float Cmax[16];
#pragma unroll
  for (int g = 0; g < 16; ++g) Cmax[g] = -3.4e38f;

  // pass 1: x_enc MLP + scores, store C rows, track Cmax
  for (int p = s; p < e; ++p) {
    const float* hp = h + (size_t)p * 32;
    float t1[32];
#pragma unroll
    for (int c = 0; c < 32; ++c) {
      float acc = 0.f;
#pragma unroll
      for (int i = 0; i < 32; ++i) acc += hp[i] * sA[i * 32 + c];
#pragma unroll
      for (int i = 0; i < 32; ++i) acc += hs[i] * sA[(32 + i) * 32 + c];
      t1[c] = fmaxf(acc, 0.f);
    }
    float t2[32];
#pragma unroll
    for (int c = 0; c < 32; ++c) {
      float acc = 0.f;
#pragma unroll
      for (int i = 0; i < 32; ++i) acc += t1[i] * sB[i * 32 + c];
      t2[c] = fmaxf(acc, 0.f);
    }
    float* crow = Cws + (size_t)p * 16;
#pragma unroll
    for (int g = 0; g < 16; ++g) {
      float cv = sbs[g];
#pragma unroll
      for (int i = 0; i < 32; ++i) cv += t2[i] * sS[i * 16 + g];
      crow[g] = cv;
      Cmax[g] = fmaxf(Cmax[g], cv);
    }
  }

  const float inv = rsqrtf((float)cnt);   // group_scaling 1/sqrt(count)

  // pass 2: denom per group
  float denom[16];
#pragma unroll
  for (int g = 0; g < 16; ++g) denom[g] = 0.f;
  for (int p = s; p < e; ++p) {
    const float* crow = Cws + (size_t)p * 16;
#pragma unroll
    for (int g = 0; g < 16; ++g) denom[g] += __expf((crow[g] - Cmax[g]) * inv);
  }
#pragma unroll
  for (int g = 0; g < 16; ++g) denom[g] += 1e-12f;

  float gate[16];
#pragma unroll
  for (int g = 0; g < 16; ++g) {
    float z = sgw[g] * Cmax[g] + sgb[g];
    gate[g] = tanhf(fmaxf(z, 0.f));
  }

  // pass 3: pooled sum, 32-channel chunks ( sum(x*e)/denom == sum(x*e/denom) )
  for (int cb = 0; cb < 4; ++cb) {
    float S[32];
#pragma unroll
    for (int c = 0; c < 32; ++c) S[c] = 0.f;
    for (int p = s; p < e; ++p) {
      const float* crow = Cws + (size_t)p * 16;
      float w16[16];
#pragma unroll
      for (int g = 0; g < 16; ++g)
        w16[g] = __expf((crow[g] - Cmax[g]) * inv) / denom[g];
      const unsigned short* xr = xme + (size_t)p * 128 + cb * 32;
#pragma unroll
      for (int c = 0; c < 32; ++c) {
        int ch = cb * 32 + c;
        S[c] += bf16_to_f32(xr[c]) * w16[ch >> 3];   // CHAN2GROUP = ch/8
      }
    }
#pragma unroll
    for (int c = 0; c < 32; ++c) {
      int ch = cb * 32 + c;
      orow[ch] = S[c] * gate[ch >> 3];
    }
  }
}

// ---------------------------------------------------------------------------
// Launcher
// ---------------------------------------------------------------------------
extern "C" void kernel_launch(void* const* d_in, const int* in_sizes, int n_in,
                              void* d_out, int out_size, void* d_ws, size_t ws_size,
                              hipStream_t stream) {
  const float* x_mod  = (const float*)d_in[0];
  const float* x_map  = (const float*)d_in[1];
  const int*   csr    = (const int*)d_in[2];
  const float* W_e1a  = (const float*)d_in[3];
  const float* W_e1b  = (const float*)d_in[4];
  const float* W_seta = (const float*)d_in[5];
  const float* W_setb = (const float*)d_in[6];
  const float* W_e2a  = (const float*)d_in[7];
  const float* W_e2b  = (const float*)d_in[8];
  const float* W_mod1 = (const float*)d_in[9];
  const float* W_mod2 = (const float*)d_in[10];
  const float* W_score= (const float*)d_in[11];
  const float* b_score= (const float*)d_in[12];
  const float* g_w    = (const float*)d_in[13];
  const float* g_b    = (const float*)d_in[14];

  const int V = in_sizes[1] / 8;     // x_map is [V,8]
  const int N = in_sizes[2] - 1;     // csr_idx is [N+1]

  char* ws = (char*)d_ws;
  size_t off = 0;
  unsigned short* w1bf = (unsigned short*)(ws + off); off += (size_t)128 * 128 * 2;
  unsigned short* w2bf = (unsigned short*)(ws + off); off += (size_t)128 * 128 * 2;
  float* h_ws    = (float*)(ws + off); off += (size_t)V * 32 * 4;
  float* hset_ws = (float*)(ws + off); off += (size_t)N * 32 * 4;
  float* C_ws    = (float*)(ws + off); off += (size_t)V * 16 * 4;
  unsigned short* xme_ws = (unsigned short*)(ws + off); off += (size_t)V * 128 * 2;
  (void)ws_size; (void)out_size; (void)n_in;

  float* out = (float*)d_out;

  k_wcvt<<<64, 256, 0, stream>>>(W_mod1, W_mod2, w1bf, w2bf);
  k_enc1<<<(V + 255) / 256, 256, 0, stream>>>(x_map, W_e1a, W_e1b, h_ws, V);
  k_modmlp<<<(V + 127) / 128, 256, 0, stream>>>(x_mod, w1bf, w2bf, xme_ws, V);
  k_set<<<(N + 255) / 256, 256, 0, stream>>>(csr, h_ws, W_seta, W_setb, hset_ws, N);
  k_pool<<<(N + 255) / 256, 256, 0, stream>>>(csr, h_ws, hset_ws, xme_ws,
                                              W_e2a, W_e2b, W_score, b_score,
                                              g_w, g_b, C_ws, out, N);
}
